// DynamicGraphTopologyLearner_33827162423951
// MI455X (gfx1250) — matrix operand order repaired and checked by reading, not compiled, not run on hardware
//
#include <hip/hip_runtime.h>
#include <hip/hip_bf16.h>
#include <math.h>

typedef __attribute__((ext_vector_type(16))) _Float16 v16h;
typedef __attribute__((ext_vector_type(8)))  float    v8f;

#define N_NODES 1024
#define F_DIM   128
#define H_DIM   64
#define HH      32
#define LN_EPSF 1e-5f
#define INV_TEMP 10.0f
#define SPARSE_TH 0.1f

// pack two f32 -> one dword of two f16 (lo, hi)
__device__ __forceinline__ unsigned pack2h(float a, float b) {
  union { _Float16 h[2]; unsigned u; } t;
  t.h[0] = (_Float16)a; t.h[1] = (_Float16)b;
  return t.u;
}

// K offset inside a 32-wide K chunk for the f16 A-matrix fragment:
// lanes 0-15 hold K {0-7,16-23}, lanes 16-31 hold K {8-15,24-31};
// vgpr v<4 -> first 8-group, v>=4 -> second; 2 halves per vgpr.
__device__ __forceinline__ int a_koff(int v, int hi, int h) {
  return ((v < 4) ? (hi * 8) : (16 + hi * 8)) + ((v & 3) * 2) + h;
}

#define WMMA_F16(A, B, C) \
  __builtin_amdgcn_wmma_f32_16x16x32_f16(false, (A), false, (B), (short)0, (C), false, false)

// CDNA5 async global->LDS copy (ASYNCcnt-tracked DMA, no VGPR data movement)
__device__ __forceinline__ void async_copy_b128(unsigned lds_off, const void* gaddr) {
  asm volatile("global_load_async_to_lds_b128 %0, %1, off"
               :: "v"(lds_off), "v"(gaddr) : "memory");
}
__device__ __forceinline__ void wait_async_le8() {
  asm volatile("s_wait_asynccnt 0x8" ::: "memory");
}
__device__ __forceinline__ void wait_async_0() {
  asm volatile("s_wait_asynccnt 0x0" ::: "memory");
}

// ---------------------------------------------------------------------------
// Kernel 1: e = LN(ReLU(X@W0+b0));  pa = e@W1[:H];  pb = e@W1[H:]
// one block = 16 rows, 4 waves; wave w owns output column tile w (16 cols)
// ---------------------------------------------------------------------------
__global__ void __launch_bounds__(128)
k1_embed(const float* __restrict__ X,  const float* __restrict__ W0,
         const float* __restrict__ b0, const float* __restrict__ g0,
         const float* __restrict__ be0,const float* __restrict__ W1,
         float* __restrict__ pa, float* __restrict__ pb) {
  __shared__ _Float16 lds_x[16 * F_DIM];     // 4 KB, X tile as f16
  __shared__ float    lds_eraw[16 * H_DIM];  // 4 KB
  __shared__ float    lds_mu[16], lds_rs[16];
  __shared__ _Float16 lds_eh[16 * H_DIM];    // 2 KB, normalized e as f16

  const int t   = threadIdx.x;
  const int wv  = t >> 5;
  const int ln  = t & 31;
  const int hi  = ln >> 4;
  const int l15 = ln & 15;
  const int i0  = blockIdx.x * 16;

  // stage X tile (16x128) into LDS as f16
  for (int p = 0; p < 16; ++p) {
    int idx = t + 128 * p;
    int r = idx >> 7, c = idx & 127;
    lds_x[idx] = (_Float16)X[(i0 + r) * F_DIM + c];
  }
  __syncthreads();

  // GEMM1: acc[16x16] = X_tile @ W0[:, 16w..16w+15], K = 128 (4 chunks)
  v8f acc = {};
  const int col = 16 * wv + l15;
  for (int kc = 0; kc < 4; ++kc) {
    union { v16h h; unsigned u[8]; } A, B;
    #pragma unroll
    for (int v = 0; v < 8; ++v) {
      int k  = 32 * kc + a_koff(v, hi, 0);
      A.u[v] = *(const unsigned*)&lds_x[l15 * F_DIM + k];
      int bk = 32 * kc + a_koff(v, hi, 0);         // B: lanes 0-15 K {0-7,16-23}, lanes 16-31 K {8-15,24-31}
      B.u[v] = pack2h(W0[bk * H_DIM + col], W0[(bk + 1) * H_DIM + col]);
    }
    acc = WMMA_F16(A.h, B.h, acc);
  }
  // bias + ReLU, park raw pre-LN values in LDS
  {
    float bias = b0[col];
    #pragma unroll
    for (int c = 0; c < 8; ++c) {
      int row = c + 8 * hi;
      float v = acc[c] + bias;
      lds_eraw[row * H_DIM + col] = v > 0.f ? v : 0.f;
    }
  }
  __syncthreads();

  // per-row LayerNorm stats (16 rows, 1 thread each; tiny)
  if (t < 16) {
    float s = 0.f, s2 = 0.f;
    for (int c = 0; c < H_DIM; ++c) { float v = lds_eraw[t * H_DIM + c]; s += v; s2 += v * v; }
    float mu  = s * (1.f / 64.f);
    float var = s2 * (1.f / 64.f) - mu * mu;
    lds_mu[t] = mu; lds_rs[t] = rsqrtf(var + LN_EPSF);
  }
  __syncthreads();
  for (int p = 0; p < 8; ++p) {
    int idx = t + 128 * p;
    int r = idx >> 6, c = idx & 63;
    float v = (lds_eraw[idx] - lds_mu[r]) * lds_rs[r] * g0[c] + be0[c];
    lds_eh[idx] = (_Float16)v;
  }
  __syncthreads();

  // GEMM2: pa/pb tiles = e_tile @ W1a / W1b, K = 64 (2 chunks)
  v8f aa = {}, ab = {};
  for (int kc = 0; kc < 2; ++kc) {
    union { v16h h; unsigned u[8]; } A, Ba, Bb;
    #pragma unroll
    for (int v = 0; v < 8; ++v) {
      int k  = 32 * kc + a_koff(v, hi, 0);
      A.u[v] = *(const unsigned*)&lds_eh[l15 * H_DIM + k];
      int bk = 32 * kc + a_koff(v, hi, 0);
      Ba.u[v] = pack2h(W1[bk * H_DIM + col],        W1[(bk + 1) * H_DIM + col]);
      Bb.u[v] = pack2h(W1[(64 + bk) * H_DIM + col], W1[(64 + bk + 1) * H_DIM + col]);
    }
    aa = WMMA_F16(A.h, Ba.h, aa);
    ab = WMMA_F16(A.h, Bb.h, ab);
  }
  #pragma unroll
  for (int c = 0; c < 8; ++c) {
    int row = c + 8 * hi;
    pa[(i0 + row) * H_DIM + col] = aa[c];
    pb[(i0 + row) * H_DIM + col] = ab[c];
  }
}

// ---------------------------------------------------------------------------
// Kernel 2: per pair (i,j): p = LN(ReLU(pa[i]+pb[j]+b1));
//           s[i,j] = sigmoid(ReLU(p@W2+b2)@W3 + b3)
// grid (N, 2), block 256 (8 waves); wave w handles 4 j-tiles of 16 pairs,
// pb tiles double-buffered in LDS via async global->LDS DMA.
// ---------------------------------------------------------------------------
__global__ void __launch_bounds__(256)
k2_pair(const float* __restrict__ pa, const float* __restrict__ pb,
        const float* __restrict__ b1, const float* __restrict__ g1,
        const float* __restrict__ be1,const float* __restrict__ W2,
        const float* __restrict__ b2, const float* __restrict__ W3,
        const float* __restrict__ b3, float* __restrict__ s_out) {
  __shared__ unsigned w2p[32 * 32];   // W2 as packed f16 pairs in B-fragment order [kpair][col]
  __shared__ float pab[H_DIM];        // pa[i] + b1 (folded once per block)
  __shared__ float lg1[H_DIM], lbe1[H_DIM];
  __shared__ float lb2[HH], lw3[HH];
  __shared__ __align__(16) float pbbuf[8][2][16 * H_DIM];  // per-wave double buffer (64 KB)

  const int t   = threadIdx.x;
  const int wv  = t >> 5;
  const int ln  = t & 31;
  const int hi  = ln >> 4;
  const int l15 = ln & 15;
  const int i   = blockIdx.x;
  const int jt_base = (blockIdx.y * 8 + wv) * 4;   // 4 j-tiles per wave

  for (int p = t; p < 1024; p += 256) {
    int kp = p >> 5, c = p & 31;
    w2p[p] = pack2h(W2[(2 * kp) * HH + c], W2[(2 * kp + 1) * HH + c]);
  }
  if (t < 64)      { pab[t] = pa[i * H_DIM + t] + b1[t]; lg1[t] = g1[t]; lbe1[t] = be1[t]; }
  else if (t < 96) { lb2[t - 64] = b2[t - 64]; lw3[t - 64] = W3[t - 64]; }
  __syncthreads();

  // B fragments (W2 in f16): built once per wave, reused for 4 j-tiles
  union { v16h h; unsigned u[8]; } B00, B01, B10, B11;
  #pragma unroll
  for (int v = 0; v < 8; ++v) {
    int kp = a_koff(v, hi, 0) >> 1;          // K = a_koff(v, hi, 0) within chunk
    B00.u[v] = w2p[kp * 32 + l15];
    B01.u[v] = w2p[kp * 32 + 16 + l15];
    B10.u[v] = w2p[(16 + kp) * 32 + l15];
    B11.u[v] = w2p[(16 + kp) * 32 + 16 + l15];
  }

  // wave-private LDS offsets of the two pb tile buffers
  const unsigned ldsb[2] = {
    (unsigned)(unsigned long long)(&pbbuf[wv][0][0]),
    (unsigned)(unsigned long long)(&pbbuf[wv][1][0])
  };
  // stage one 16x64 f32 pb tile (4 KB) with 8 async b128 copies (per wave)
  auto stage = [&](int jt, int buf) {
    const char* src = (const char*)(pb + jt * 16 * H_DIM);
    #pragma unroll
    for (int c = 0; c < 8; ++c) {
      unsigned off = (unsigned)((c * 32 + ln) * 16);
      async_copy_b128(ldsb[buf] + off, src + off);
    }
  };

  stage(jt_base, 0);

  const float b2a = lb2[l15], b2b = lb2[16 + l15];
  const float w3a = lw3[l15], w3b = lw3[16 + l15];
  const float bias3 = b3[0];

  for (int it = 0; it < 4; ++it) {
    if (it < 3) { stage(jt_base + it + 1, (it + 1) & 1); wait_async_le8(); }
    else        { wait_async_0(); }
    const float* __restrict__ pbl = &pbbuf[wv][it & 1][l15 * H_DIM];

    // compute the 32 p-values this lane owns, directly in A-fragment order
    float pv[32];
    float sum = 0.f, sq = 0.f;
    #pragma unroll
    for (int chunk = 0; chunk < 2; ++chunk)
      #pragma unroll
      for (int e = 0; e < 16; ++e) {
        int v = e >> 1, h = e & 1;
        int k = 32 * chunk + a_koff(v, hi, h);
        float x = pab[k] + pbl[k];
        x = x > 0.f ? x : 0.f;
        pv[chunk * 16 + e] = x;
        sum += x; sq += x * x;
      }
    // row r=l15 is split across lanes ln and ln^16: one exchange completes LN stats
    sum += __shfl_xor(sum, 16);
    sq  += __shfl_xor(sq, 16);
    float mu   = sum * (1.f / 64.f);
    float rstd = rsqrtf(sq * (1.f / 64.f) - mu * mu + LN_EPSF);

    union { v16h h; unsigned u[8]; } A0, A1;
    #pragma unroll
    for (int e = 0; e < 16; ++e) {
      int v = e >> 1, h = e & 1;
      int k0 = a_koff(v, hi, h);
      A0.h[e] = (_Float16)((pv[e]      - mu) * rstd * lg1[k0]      + lbe1[k0]);
      A1.h[e] = (_Float16)((pv[16 + e] - mu) * rstd * lg1[32 + k0] + lbe1[32 + k0]);
    }

    v8f acc0 = {}, acc1 = {};
    acc0 = WMMA_F16(A0.h, B00.h, acc0);
    acc0 = WMMA_F16(A1.h, B10.h, acc0);
    acc1 = WMMA_F16(A0.h, B01.h, acc1);
    acc1 = WMMA_F16(A1.h, B11.h, acc1);

    // epilogue: q = ReLU(acc + b2); scalar = q . W3; sigmoid
    const int j0 = (jt_base + it) * 16;
    #pragma unroll
    for (int c = 0; c < 8; ++c) {
      float q0 = acc0[c] + b2a; q0 = q0 > 0.f ? q0 : 0.f;
      float q1 = acc1[c] + b2b; q1 = q1 > 0.f ? q1 : 0.f;
      float ts = q0 * w3a + q1 * w3b;          // partial over this lane's column
      ts += __shfl_xor(ts, 1);
      ts += __shfl_xor(ts, 2);
      ts += __shfl_xor(ts, 4);
      ts += __shfl_xor(ts, 8);                 // full sum over 16 cols (stays within half)
      if (l15 == c) {
        float sv = 1.f / (1.f + __expf(-(ts + bias3)));
        s_out[i * N_NODES + j0 + c + 8 * hi] = sv;   // D row = c + 8*hi
      }
    }
  }
}

// ---------------------------------------------------------------------------
// Kernel 3: edge_weights = (triu(s,1)+triu(s,1)^T)/T ; mask = ew > 0.1
// ---------------------------------------------------------------------------
__global__ void __launch_bounds__(256)
k3_sym(const float* __restrict__ s, float* __restrict__ ew, float* __restrict__ mask) {
  int idx = blockIdx.x * 256 + threadIdx.x;
  int i = idx >> 10, j = idx & 1023;
  float v = 0.f;
  if (i < j)      v = s[i * N_NODES + j];
  else if (i > j) v = s[j * N_NODES + i];
  v *= INV_TEMP;
  ew[idx]   = v;
  mask[idx] = (v > SPARSE_TH) ? 1.f : 0.f;
}

extern "C" void kernel_launch(void* const* d_in, const int* in_sizes, int n_in,
                              void* d_out, int out_size, void* d_ws, size_t ws_size,
                              hipStream_t stream) {
  const float* X   = (const float*)d_in[0];
  const float* W0  = (const float*)d_in[1];
  const float* b0  = (const float*)d_in[2];
  const float* g0  = (const float*)d_in[3];
  const float* be0 = (const float*)d_in[4];
  const float* W1  = (const float*)d_in[5];
  const float* b1  = (const float*)d_in[6];
  const float* g1  = (const float*)d_in[7];
  const float* be1 = (const float*)d_in[8];
  const float* W2  = (const float*)d_in[9];
  const float* b2  = (const float*)d_in[10];
  const float* W3  = (const float*)d_in[11];
  const float* b3  = (const float*)d_in[12];

  float* pa = (float*)d_ws;                       // [N, H]
  float* pb = pa + N_NODES * H_DIM;               // [N, H]
  float* s  = pb + N_NODES * H_DIM;               // [N, N]

  float* ew   = (float*)d_out;                    // [N, N]
  float* mask = ew + N_NODES * N_NODES;           // [N, N] as 0/1 floats

  k1_embed<<<dim3(N_NODES / 16), dim3(128), 0, stream>>>(X, W0, b0, g0, be0, W1, pa, pb);
  k2_pair <<<dim3(N_NODES, 2),   dim3(256), 0, stream>>>(pa, pb, b1, g1, be1, W2, b2, W3, b3, s);
  k3_sym  <<<dim3((N_NODES * N_NODES) / 256), dim3(256), 0, stream>>>(s, ew, mask);
}
